// AtomCrossAttEncoder_33913061769455
// MI455X (gfx1250) — matrix operand order repaired and checked by reading, not compiled, not run on hardware
//
#include <hip/hip_runtime.h>

// ---------------- model dimensions ----------------
#define TT    384
#define AA    24
#define SS    288
#define QQ    32
#define KKk   128
#define CAc   128
#define CPc   16
#define CSc   384
#define CPrc  128
#define CTc   768
#define NBc   3
#define NHc   4
#define DHc   32
#define BBc   4

#define SQn   (SS*QQ)      // 9216
#define SKn   (SS*KKk)     // 36864
#define BSQn  (BBc*SQn)    // 36864
#define BSKn  (BBc*SKn)    // 147456

typedef __attribute__((ext_vector_type(16))) __bf16 v16bf;
typedef __attribute__((ext_vector_type(4)))  __bf16 v4bf;
typedef __attribute__((ext_vector_type(8)))  float  v8f;

__device__ inline v8f wmma_bf16(v16bf a, v16bf b, v8f c) {
  return __builtin_amdgcn_wmma_f32_16x16x32_bf16(false, a, false, b, (short)0, c,
                                                 false, false);
}

// Build a 16-element bf16 fragment from two 8-element contiguous LDS runs.
__device__ inline v16bf frag_ld(const __bf16* p0, const __bf16* p1) {
  v4bf a0 = *(const v4bf*)(p0);
  v4bf a1 = *(const v4bf*)(p0 + 4);
  v4bf a2 = *(const v4bf*)(p1);
  v4bf a3 = *(const v4bf*)(p1 + 4);
  v16bf f;
  #pragma unroll
  for (int e = 0; e < 4; ++e) {
    f[e]      = a0[e];
    f[e + 4]  = a1[e];
    f[e + 8]  = a2[e];
    f[e + 12] = a3[e];
  }
  return f;
}

// =======================================================================
// Generic WMMA GEMM:  C[M,N] = EPI( PRE(A)[M,K] @ W[K,N] (+bias) )
// PRE: 0 = none, 1 = relu, 2 = rowwise LayerNorm * lnscale
// EPI: 0 = none, 1 = +bias, 2 = sigmoid(x + bias)  (bias may be null)
// block = 128 threads (4 waves); wave tile = 32 x (16*NSUB); wg = 128 x 16*NSUB
// A and W staged in LDS as bf16 (one cvt at staging); W stored transposed so
// both fragment types are contiguous 8B vector LDS loads. K % 32 == 0,
// M % 128 == 0 assumed (true for all uses).
// =======================================================================
template<int PRE, int EPI, int NSUB>
__global__ __launch_bounds__(128) void enc_gemm(
    const float* __restrict__ A, const float* __restrict__ W,
    const float* __restrict__ bias, const float* __restrict__ lnsc,
    float* __restrict__ C, int M, int K, int N)
{
  __shared__ __attribute__((aligned(16))) __bf16 As[128][36];
  __shared__ __attribute__((aligned(16))) __bf16 Wt[16*NSUB][36];
  __shared__ float mu_s[128], rs_s[128];

  const int tid  = threadIdx.x;
  const int wave = tid >> 5;
  const int lane = tid & 31;
  const int hi   = lane >> 4;
  const int ln16 = lane & 15;
  const int m_blk = blockIdx.x * 128;
  const int n_blk = blockIdx.y * 16 * NSUB;

  if constexpr (PRE == 2) {
    int r = m_blk + tid;
    float s = 0.f, s2 = 0.f;
    if (r < M) {
      const float* row = A + (size_t)r * K;
      for (int c = 0; c < K; ++c) { float v = row[c]; s += v; s2 += v * v; }
    }
    float mean = s / (float)K;
    float var  = s2 / (float)K - mean * mean;
    mu_s[tid] = mean;
    rs_s[tid] = rsqrtf(var + 1e-5f);
    __syncthreads();
  }

  v8f acc[2][NSUB];
  #pragma unroll
  for (int mi = 0; mi < 2; ++mi)
    #pragma unroll
    for (int i = 0; i < NSUB; ++i)
      #pragma unroll
      for (int r = 0; r < 8; ++r) acc[mi][i][r] = 0.f;

  for (int kk = 0; kk < K; kk += 32) {
    // stage A chunk: 128 rows x 32 K, convert to bf16 once
    for (int i = tid; i < 128 * 32; i += 128) {
      int r = i >> 5, c = i & 31;
      int gr = m_blk + r;
      float v = (gr < M) ? A[(size_t)gr * K + kk + c] : 0.f;
      if constexpr (PRE == 1) v = fmaxf(v, 0.f);
      if constexpr (PRE == 2) v = (v - mu_s[r]) * rs_s[r] * (lnsc ? lnsc[kk + c] : 1.f);
      As[r][c] = (__bf16)v;
    }
    // stage W chunk transposed: Wt[n][k]
    for (int i = tid; i < 32 * 16 * NSUB; i += 128) {
      int r = i / (16 * NSUB), c = i % (16 * NSUB);
      int gn = n_blk + c;
      Wt[c][r] = (__bf16)((gn < N) ? W[(size_t)(kk + r) * N + gn] : 0.f);
    }
    // prefetch next A chunk (gfx1250 global_prefetch_b8)
    if (kk + 32 < K) {
      int r = tid >> 5, c = tid & 31;
      __builtin_prefetch(&A[(size_t)(m_blk + r) * K + kk + 32 + c], 0, 1);
    }
    __syncthreads();

    v16bf af0, af1;
    {
      int m0 = wave * 32 + ln16;
      af0 = frag_ld(&As[m0][8 * hi],      &As[m0][16 + 8 * hi]);
      af1 = frag_ld(&As[m0 + 16][8 * hi], &As[m0 + 16][16 + 8 * hi]);
    }
    #pragma unroll
    for (int ns = 0; ns < NSUB; ++ns) {
      int ncol = ns * 16 + ln16;
      v16bf bfv = frag_ld(&Wt[ncol][8 * hi], &Wt[ncol][16 + 8 * hi]);
      acc[0][ns] = wmma_bf16(af0, bfv, acc[0][ns]);
      acc[1][ns] = wmma_bf16(af1, bfv, acc[1][ns]);
    }
    __syncthreads();
  }

  #pragma unroll
  for (int mi = 0; mi < 2; ++mi) {
    #pragma unroll
    for (int ns = 0; ns < NSUB; ++ns) {
      int gc = n_blk + ns * 16 + ln16;
      if (gc >= N) continue;
      #pragma unroll
      for (int r = 0; r < 8; ++r) {
        int gr = m_blk + wave * 32 + mi * 16 + r + 8 * hi;
        if (gr >= M) continue;
        float v = acc[mi][ns][r];
        if constexpr (EPI >= 1) { if (bias) v += bias[gc]; }
        if constexpr (EPI == 2) v = 1.f / (1.f + __expf(-v));
        C[(size_t)gr * N + gc] = v;
      }
    }
  }
}

// =======================================================================
// Atom feature embedding -> act (T*A, 128)
// =======================================================================
__global__ __launch_bounds__(128) void enc_atom_embed(
    const float* __restrict__ ref_pos, const float* __restrict__ ref_mask,
    const float* __restrict__ ref_charge, const int* __restrict__ ref_elem,
    const int* __restrict__ name_chars,
    const float* __restrict__ w_pos, const float* __restrict__ w_mask,
    const float* __restrict__ w_elem, const float* __restrict__ w_charge,
    const float* __restrict__ w_name, float* __restrict__ act)
{
  int ta = blockIdx.x, c = threadIdx.x;
  float rp0 = ref_pos[ta*3+0], rp1 = ref_pos[ta*3+1], rp2 = ref_pos[ta*3+2];
  float m = ref_mask[ta];
  float v = rp0 * w_pos[c] + rp1 * w_pos[CAc + c] + rp2 * w_pos[2*CAc + c];
  v += m * w_mask[c];
  v += w_elem[(size_t)ref_elem[ta] * CAc + c];
  v += asinhf(ref_charge[ta]) * w_charge[c];
  #pragma unroll
  for (int j = 0; j < 4; ++j)
    v += w_name[(size_t)(j * 64 + name_chars[ta*4 + j]) * CAc + c];
  act[(size_t)ta * CAc + c] = v * m;
}

// queries_single_cond + queries_mask
__global__ void enc_build_qsc(const float* __restrict__ act, const float* __restrict__ ts,
                              const int* __restrict__ t2q, const int* __restrict__ tok2q,
                              const float* __restrict__ atom_mask,
                              float* __restrict__ qsc, float* __restrict__ qmask)
{
  int i = blockIdx.x * blockDim.x + threadIdx.x;
  if (i >= SQn * CAc) return;
  int sq = i / CAc, c = i % CAc;
  int idx = t2q[sq];
  qsc[i] = act[(size_t)idx * CAc + c] + ts[(size_t)tok2q[sq] * CAc + c];
  if (c == 0) qmask[sq] = atom_mask[idx];
}

// x init: (pos_feat projection)*mask + qsc
__global__ void enc_build_x(const float* __restrict__ taa, const float* __restrict__ w_pf,
                            const int* __restrict__ t2q, const float* __restrict__ qmask,
                            const float* __restrict__ qsc, float* __restrict__ x)
{
  int i = blockIdx.x * blockDim.x + threadIdx.x;
  if (i >= BSQn * CAc) return;
  int c  = i % CAc;
  int sq = (i / CAc) % SQn;
  int bb = i / (SQn * CAc);
  int idx = t2q[sq];
  const float* p = taa + ((size_t)bb * TT * AA + idx) * 3;
  float v = p[0] * w_pf[c] + p[1] * w_pf[CAc + c] + p[2] * w_pf[2*CAc + c];
  x[i] = v * qmask[sq] + qsc[(size_t)sq * CAc + c];
}

// keys_single_cond + keys_mask (gather of qsc/qmask)
__global__ void enc_build_ksc(const float* __restrict__ qsc, const float* __restrict__ qmask,
                              const int* __restrict__ q2k,
                              float* __restrict__ ksc, float* __restrict__ kmask)
{
  int i = blockIdx.x * blockDim.x + threadIdx.x;
  if (i >= SKn * CAc) return;
  int sk = i / CAc, c = i % CAc;
  int idx = q2k[sk];
  ksc[i] = qsc[(size_t)idx * CAc + c];
  if (c == 0) kmask[sk] = qmask[idx];
}

// =======================================================================
// Pair kernel: one block per (s,q); thread = key position. Builds pair,
// runs 16x16x3 MLP in registers, writes pair + pair logits.
// =======================================================================
__global__ __launch_bounds__(128) void enc_pair(
    const float* __restrict__ rowb, const float* __restrict__ colb,
    const float* __restrict__ tp, const float* __restrict__ ref_pos,
    const int* __restrict__ uid,
    const int* __restrict__ t2q, const int* __restrict__ q2k,
    const int* __restrict__ tok2q, const int* __restrict__ tok2k,
    const float* __restrict__ wpo, const float* __restrict__ wpd,
    const float* __restrict__ wpv,
    const float* __restrict__ m1, const float* __restrict__ m2,
    const float* __restrict__ m3,
    const float* __restrict__ lnsc, const float* __restrict__ wl,
    float* __restrict__ pair_out, float* __restrict__ pl_out)
{
  __shared__ float s_m1[256], s_m2[256], s_m3[256], s_wl[192];
  __shared__ float s_row[16], s_wpo[48], s_wpd[16], s_wpv[16], s_ln[16];
  int tid = threadIdx.x;
  int sq  = blockIdx.x;
  for (int i = tid; i < 256; i += 128) { s_m1[i]=m1[i]; s_m2[i]=m2[i]; s_m3[i]=m3[i]; }
  for (int i = tid; i < 192; i += 128) s_wl[i] = wl[i];
  if (tid < 48) s_wpo[tid] = wpo[tid];
  if (tid < 16) { s_wpd[tid]=wpd[tid]; s_wpv[tid]=wpv[tid]; s_ln[tid]=lnsc[tid];
                  s_row[tid]=rowb[(size_t)sq*16 + tid]; }
  __syncthreads();

  int s  = sq / QQ;
  int qidx = t2q[sq];
  float qx = ref_pos[qidx*3+0], qy = ref_pos[qidx*3+1], qz = ref_pos[qidx*3+2];
  int quid = uid[qidx], tq = tok2q[sq];

  int kk  = tid;                 // 0..127
  int skk = s * KKk + kk;
  int kidx = q2k[skk];
  int tk = tok2k[skk];
  const float* colp = colb + (size_t)skk * 16;
  const float* tpp  = tp + ((size_t)tq * TT + tk) * 16;

  float p[16];
  #pragma unroll
  for (int j = 0; j < 16; ++j) p[j] = s_row[j] + colp[j] + tpp[j];

  float ox = qx - ref_pos[kidx*3+0];
  float oy = qy - ref_pos[kidx*3+1];
  float oz = qz - ref_pos[kidx*3+2];
  float vf = (quid == uid[kidx]) ? 1.f : 0.f;
  float invd = 1.f / (1.f + ox*ox + oy*oy + oz*oz);
  #pragma unroll
  for (int j = 0; j < 16; ++j) {
    p[j] += (ox * s_wpo[j] + oy * s_wpo[16+j] + oz * s_wpo[32+j]) * vf;
    p[j] += invd * s_wpd[j] * vf + vf * s_wpv[j];
  }

  float t[16], u[16];
  #pragma unroll
  for (int j = 0; j < 16; ++j) { float a=0;
    #pragma unroll
    for (int i2 = 0; i2 < 16; ++i2) a += fmaxf(p[i2],0.f) * s_m1[i2*16+j];
    t[j]=a; }
  #pragma unroll
  for (int j = 0; j < 16; ++j) { float a=0;
    #pragma unroll
    for (int i2 = 0; i2 < 16; ++i2) a += fmaxf(t[i2],0.f) * s_m2[i2*16+j];
    u[j]=a; }
  #pragma unroll
  for (int j = 0; j < 16; ++j) { float a=0;
    #pragma unroll
    for (int i2 = 0; i2 < 16; ++i2) a += fmaxf(u[i2],0.f) * s_m3[i2*16+j];
    p[j]=a; }

  size_t base = ((size_t)sq * KKk + kk) * 16;
  #pragma unroll
  for (int j = 0; j < 16; ++j) pair_out[base + j] = p[j];

  float mu = 0.f;
  #pragma unroll
  for (int j = 0; j < 16; ++j) mu += p[j];
  mu *= (1.f/16.f);
  float var = 0.f;
  #pragma unroll
  for (int j = 0; j < 16; ++j) { float d = p[j]-mu; var += d*d; }
  var *= (1.f/16.f);
  float rs = rsqrtf(var + 1e-5f);
  float xn[16];
  #pragma unroll
  for (int j = 0; j < 16; ++j) xn[j] = (p[j]-mu) * rs * s_ln[j];
  size_t pb = ((size_t)sq * KKk + kk) * 12;
  #pragma unroll
  for (int n = 0; n < 12; ++n) { float a=0;
    #pragma unroll
    for (int j = 0; j < 16; ++j) a += xn[j] * s_wl[j*12+n];
    pl_out[pb + n] = a; }
}

// =======================================================================
// adaLN combine: out = cs * LN(x_row) + cb   (one block per row, 128 thr)
// =======================================================================
__global__ __launch_bounds__(128) void enc_adaln_q(
    const float* __restrict__ x, const float* __restrict__ cs,
    const float* __restrict__ cb, float* __restrict__ out)
{
  __shared__ float red[128];
  int row = blockIdx.x, c = threadIdx.x;
  int sq = row % SQn;
  float v = x[(size_t)row * CAc + c];
  red[c] = v; __syncthreads();
  for (int s2 = 64; s2 > 0; s2 >>= 1) { if (c < s2) red[c] += red[c+s2]; __syncthreads(); }
  float mu = red[0] * (1.f/128.f); __syncthreads();
  float d = v - mu;
  red[c] = d*d; __syncthreads();
  for (int s2 = 64; s2 > 0; s2 >>= 1) { if (c < s2) red[c] += red[c+s2]; __syncthreads(); }
  float var = red[0] * (1.f/128.f);
  float xn = d * rsqrtf(var + 1e-5f);
  out[(size_t)row * CAc + c] = cs[(size_t)sq * CAc + c] * xn + cb[(size_t)sq * CAc + c];
}

__global__ __launch_bounds__(128) void enc_adaln_k(
    const float* __restrict__ x, const int* __restrict__ q2k,
    const float* __restrict__ cs, const float* __restrict__ cb,
    float* __restrict__ out)
{
  __shared__ float red[128];
  int row = blockIdx.x, c = threadIdx.x;
  int bb = row / SKn, sk = row % SKn;
  size_t src = ((size_t)bb * SQn + q2k[sk]) * CAc;
  float v = x[src + c];
  red[c] = v; __syncthreads();
  for (int s2 = 64; s2 > 0; s2 >>= 1) { if (c < s2) red[c] += red[c+s2]; __syncthreads(); }
  float mu = red[0] * (1.f/128.f); __syncthreads();
  float d = v - mu;
  red[c] = d*d; __syncthreads();
  for (int s2 = 64; s2 > 0; s2 >>= 1) { if (c < s2) red[c] += red[c+s2]; __syncthreads(); }
  float var = red[0] * (1.f/128.f);
  float xn = d * rsqrtf(var + 1e-5f);
  out[(size_t)row * CAc + c] = cs[(size_t)sk * CAc + c] * xn + cb[(size_t)sk * CAc + c];
}

// =======================================================================
// Attention core: one block per (s, h, b). QK^T WMMA -> softmax -> PV WMMA
// Q/K staged bf16; V staged transposed bf16 so PV B-fragments are contiguous.
// =======================================================================
__global__ __launch_bounds__(128) void enc_attn(
    const float* __restrict__ q, const float* __restrict__ k,
    const float* __restrict__ v, const float* __restrict__ pl,
    const float* __restrict__ kmask, float* __restrict__ o, int blk)
{
  __shared__ __attribute__((aligned(16))) __bf16 sQ[32][36];
  __shared__ __attribute__((aligned(16))) __bf16 sK[128][36];
  __shared__ __attribute__((aligned(16))) __bf16 sVt[32][136];   // [ch][key]
  __shared__ float sL[32][132];
  int s = blockIdx.x, h = blockIdx.y, bb = blockIdx.z;
  int tid = threadIdx.x, wave = tid >> 5, lane = tid & 31;
  int hi = lane >> 4, ln16 = lane & 15;

  size_t qbase = ((size_t)(bb * SS + s) * QQ)  * CAc + h * DHc;
  size_t kbase = ((size_t)(bb * SS + s) * KKk) * CAc + h * DHc;
  for (int i = tid; i < 32*32; i += 128) {
    int r = i >> 5, c = i & 31;
    sQ[r][c] = (__bf16)q[qbase + (size_t)r * CAc + c];
  }
  for (int i = tid; i < 128*32; i += 128) {
    int r = i >> 5, c = i & 31;
    sK[r][c]  = (__bf16)k[kbase + (size_t)r * CAc + c];
    sVt[c][r] = (__bf16)v[kbase + (size_t)r * CAc + c];
  }
  __syncthreads();

  const float scale = 0.17677669529663687f;   // 1/sqrt(32)
  for (int t = wave; t < 16; t += 4) {
    int mt = t >> 3, nt = t & 7;
    int qrow = mt * 16 + ln16;
    int key  = nt * 16 + ln16;
    v16bf af = frag_ld(&sQ[qrow][8*hi], &sQ[qrow][16 + 8*hi]);
    v16bf bfv = frag_ld(&sK[key][8*hi], &sK[key][16 + 8*hi]);
    v8f acc;
    #pragma unroll
    for (int r = 0; r < 8; ++r) acc[r] = 0.f;
    acc = wmma_bf16(af, bfv, acc);
    #pragma unroll
    for (int r = 0; r < 8; ++r)
      sL[mt*16 + r + 8*hi][nt*16 + ln16] = acc[r] * scale;
  }
  __syncthreads();

  if (tid < 32) {
    int qrow = tid;
    size_t plb = ((size_t)(s * QQ + qrow) * KKk) * 12 + blk * NHc + h;
    float mx = -1e30f;
    for (int kk = 0; kk < KKk; ++kk) {
      float val = sL[qrow][kk] + pl[plb + (size_t)kk * 12]
                + (kmask[s * KKk + kk] > 0.5f ? 0.f : -1e9f);
      sL[qrow][kk] = val;
      mx = fmaxf(mx, val);
    }
    float sum = 0.f;
    for (int kk = 0; kk < KKk; ++kk) {
      float e = __expf(sL[qrow][kk] - mx);
      sL[qrow][kk] = e; sum += e;
    }
    float inv = 1.f / sum;
    for (int kk = 0; kk < KKk; ++kk) sL[qrow][kk] *= inv;
  }
  __syncthreads();

  int mt = wave >> 1, nt = wave & 1;
  int nch = nt * 16 + ln16;
  v8f acc;
  #pragma unroll
  for (int r = 0; r < 8; ++r) acc[r] = 0.f;
  #pragma unroll
  for (int kc = 0; kc < 4; ++kc) {
    v16bf af, bfv;
    int qrow = mt * 16 + ln16;
    #pragma unroll
    for (int e = 0; e < 8; ++e)  af[e] = (__bf16)sL[qrow][kc*32 + 8*hi + e];
    #pragma unroll
    for (int e = 8; e < 16; ++e) af[e] = (__bf16)sL[qrow][kc*32 + 16 + 8*hi + (e-8)];
    bfv = frag_ld(&sVt[nch][kc*32 + 8*hi], &sVt[nch][kc*32 + 16 + 8*hi]);
    acc = wmma_bf16(af, bfv, acc);
  }
  size_t ob = ((size_t)(bb * SS + s) * QQ) * CAc + h * DHc;
  #pragma unroll
  for (int r = 0; r < 8; ++r)
    o[ob + (size_t)(mt*16 + r + 8*hi) * CAc + nt*16 + ln16] = acc[r];
}

// ---------------- small elementwise kernels ----------------
__global__ void enc_mul_inplace(float* __restrict__ a, const float* __restrict__ b, int n) {
  int i = blockIdx.x * blockDim.x + threadIdx.x;
  if (i < n) a[i] *= b[i];
}
__global__ void enc_addgate(float* __restrict__ x, const float* __restrict__ add,
                            const float* __restrict__ gate, int n, int gmod) {
  int i = blockIdx.x * blockDim.x + threadIdx.x;
  if (i < n) x[i] += add[i] * gate[i % gmod];
}
__global__ void enc_swiglu(const float* __restrict__ a, float* __restrict__ h, int n) {
  int i = blockIdx.x * blockDim.x + threadIdx.x;
  if (i >= n) return;
  int row = i >> 8, j = i & 255;
  float a1 = a[(size_t)row * 512 + j];
  float a2 = a[(size_t)row * 512 + 256 + j];
  h[i] = a1 / (1.f + __expf(-a1)) * a2;
}
__global__ void enc_mask_skip(float* __restrict__ x, const float* __restrict__ qmask,
                              float* __restrict__ skip, int n) {
  int i = blockIdx.x * blockDim.x + threadIdx.x;
  if (i >= n) return;
  int sq = (i / CAc) % SQn;
  float v = x[i] * qmask[sq];
  x[i] = v; skip[i] = v;
}
__global__ __launch_bounds__(256) void enc_token_aggr(
    const float* __restrict__ feat, const float* __restrict__ atom_mask,
    float* __restrict__ out) {
  int bb = blockIdx.x / TT, t = blockIdx.x % TT;
  for (int c = threadIdx.x; c < CTc; c += 256) {
    float num = 0.f, den = 1e-10f;
    for (int a = 0; a < AA; ++a) {
      float m = atom_mask[t * AA + a];
      float f = feat[((size_t)bb * SQn + t * AA + a) * CTc + c];
      num += fmaxf(f, 0.f) * m;
      den += m;
    }
    out[((size_t)bb * TT + t) * CTc + c] = num / den;
  }
}

// =======================================================================
// Host side
// =======================================================================
static inline dim3 g1(int n, int b) { return dim3((n + b - 1) / b); }

extern "C" void kernel_launch(void* const* d_in, const int* in_sizes, int n_in,
                              void* d_out, int out_size, void* d_ws, size_t ws_size,
                              hipStream_t stream) {
  (void)in_sizes; (void)n_in; (void)out_size; (void)ws_size;
  // ---- tensor inputs (setup_inputs() order) ----
  const float* taa      = (const float*)d_in[0];
  const float* tsc      = (const float*)d_in[1];
  const float* tpc      = (const float*)d_in[2];
  const float* ref_pos  = (const float*)d_in[3];
  const float* ref_mask = (const float*)d_in[4];
  const float* ref_chg  = (const float*)d_in[5];
  const float* atom_mask= (const float*)d_in[6];
  const int*   ref_elem = (const int*)d_in[7];
  const int*   namechar = (const int*)d_in[8];
  const int*   uid      = (const int*)d_in[9];
  const int*   t2q      = (const int*)d_in[10];
  const int*   q2k      = (const int*)d_in[11];
  const int*   tok2q    = (const int*)d_in[12];
  const int*   tok2k    = (const int*)d_in[13];
  // d_in[14] = q2ta_idx (identity; unused)
  const int PB = 15;   // params dict, insertion order
  const float* P[47];
  for (int i = 0; i < 47; ++i) P[i] = (const float*)d_in[PB + i];
  const float *w_ref_pos=P[0], *w_ref_mask=P[1], *w_ref_elem=P[2], *w_ref_chg=P[3],
              *w_ref_name=P[4], *w_row=P[5], *w_col=P[6], *w_po=P[7], *w_pd=P[8],
              *w_pv=P[9], *mlp1=P[10], *mlp2=P[11], *mlp3=P[12], *w_pf=P[13],
              *ln_ts=P[14], *w_tsngl=P[15], *ln_tp=P[16], *w_tpair=P[17],
              *ln_pair=P[18], *w_plog=P[19], *w_aggr=P[20],
              *qln=P[21], *q_ws=P[22], *q_bs=P[23], *q_wb=P[24],
              *kln=P[25], *k_ws=P[26], *k_bs=P[27], *k_wb=P[28],
              *wq=P[29], *bq=P[30], *wk=P[31], *wv=P[32], *wg=P[33], *wo=P[34],
              *zln=P[35], *wz=P[36], *bz=P[37],
              *tln=P[38], *t_ws=P[39], *t_bs=P[40], *t_wb=P[41],
              *w1=P[42], *w2=P[43], *tzln=P[44], *twz=P[45], *tbz=P[46];

  // ---- d_out regions (f32, reference return order) ----
  float* out = (float*)d_out;
  float* o_tok   = out;                         // 4*384*768
  float* o_skip  = out + 1179648;               // 4*288*32*128
  float* o_qmask = out + 5898240;               // 9216
  float* o_qsc   = out + 5907456;               // 9216*128
  float* o_kmask = out + 7087104;               // 36864
  float* o_ksc   = out + 7123968;               // 36864*128
  float* o_pair  = out + 11842560;              // 9216*128*16

  // ---- workspace layout (floats) ----
  float* ws = (float*)d_ws;
  float* f_act  = ws;                     // 1179648
  float* f_ts   = f_act  + 1179648;       // 49152
  float* f_row  = f_ts   + 49152;         // 147456
  float* f_col  = f_row  + 147456;        // 589824
  float* f_tp   = f_col  + 589824;        // 2359296
  float* f_pl   = f_tp   + 2359296;       // 14155776
  float* f_x    = f_pl   + 14155776;      // 4718592
  float* f_qn   = f_x    + 4718592;       // 4718592 (qn / tn)
  float* f_kn   = f_qn   + 4718592;       // 18874368
  float* f_q    = f_kn   + 18874368;      // 4718592 (q / gate g)
  float* f_k    = f_q    + 4718592;       // 18874368 (k / w1-act / feat lo)
  float* f_v    = f_k    + 18874368;      // 18874368 (v / swiglu-h / feat hi)
  float* f_o    = f_v    + 18874368;      // 4718592 (attn out / h2)
  float* f_o2   = f_o    + 4718592;       // 4718592
  float* f_cs   = f_o2   + 4718592;       // 4718592
  float* f_cb   = f_cs   + 4718592;       // 4718592
  float* f_gate = f_cb   + 4718592;       // 1179648

  const int TH = 256;

  // 1) atom embedding
  enc_atom_embed<<<TT*AA, 128, 0, stream>>>(ref_pos, ref_mask, ref_chg, ref_elem,
      namechar, w_ref_pos, w_ref_mask, w_ref_elem, w_ref_chg, w_ref_name, f_act);

  // 2) ts = LN(trunk_single)*scale @ w_trunk_single  (384x384x128)
  enc_gemm<2,0,4><<<dim3(TT/128, CAc/64), 128, 0, stream>>>(
      tsc, w_tsngl, nullptr, ln_ts, f_ts, TT, CSc, CAc);

  // 3) queries_single_cond + mask
  enc_build_qsc<<<g1(SQn*CAc, TH), TH, 0, stream>>>(f_act, f_ts, t2q, tok2q,
      atom_mask, o_qsc, o_qmask);

  // 4) x init
  enc_build_x<<<g1(BSQn*CAc, TH), TH, 0, stream>>>(taa, w_pf, t2q, o_qmask, o_qsc, f_x);

  // 5) keys_single_cond + mask
  enc_build_ksc<<<g1(SKn*CAc, TH), TH, 0, stream>>>(o_qsc, o_qmask, q2k, o_ksc, o_kmask);

  // 6) row = relu(qsc)@w_row ; col = relu(ksc)@w_col  (N=16)
  enc_gemm<1,0,1><<<dim3(SQn/128, 1), 128, 0, stream>>>(o_qsc, w_row, nullptr, nullptr,
      f_row, SQn, CAc, CPc);
  enc_gemm<1,0,1><<<dim3(SKn/128, 1), 128, 0, stream>>>(o_ksc, w_col, nullptr, nullptr,
      f_col, SKn, CAc, CPc);

  // 7) tp = LN(trunk_pair)*scale @ w_trunk_pair  (147456x128x16)
  enc_gemm<2,0,1><<<dim3(TT*TT/128, 1), 128, 0, stream>>>(tpc, w_tpair, nullptr, ln_tp,
      f_tp, TT*TT, CPrc, CPc);

  // 8) pair + logits
  enc_pair<<<SQn, 128, 0, stream>>>(f_row, f_col, f_tp, ref_pos, uid, t2q, q2k,
      tok2q, tok2k, w_po, w_pd, w_pv, mlp1, mlp2, mlp3, ln_pair, w_plog,
      o_pair, f_pl);

  // 9) attention blocks
  for (int b = 0; b < NBc; ++b) {
    const size_t WM = (size_t)CAc * CAc;     // 16384
    // adaLN(q): cs = sigmoid(LN(qsc)*qln @ q_ws + q_bs); cb = LN(qsc)*qln @ q_wb
    enc_gemm<2,2,4><<<dim3(SQn/128, 2), 128, 0, stream>>>(o_qsc, q_ws + b*WM,
        q_bs + b*CAc, qln + b*CAc, f_cs, SQn, CAc, CAc);
    enc_gemm<2,0,4><<<dim3(SQn/128, 2), 128, 0, stream>>>(o_qsc, q_wb + b*WM,
        nullptr, qln + b*CAc, f_cb, SQn, CAc, CAc);
    enc_adaln_q<<<BSQn, 128, 0, stream>>>(f_x, f_cs, f_cb, f_qn);
    // adaLN(k) with gather of x
    enc_gemm<2,2,4><<<dim3(SKn/128, 2), 128, 0, stream>>>(o_ksc, k_ws + b*WM,
        k_bs + b*CAc, kln + b*CAc, f_cs, SKn, CAc, CAc);
    enc_gemm<2,0,4><<<dim3(SKn/128, 2), 128, 0, stream>>>(o_ksc, k_wb + b*WM,
        nullptr, kln + b*CAc, f_cb, SKn, CAc, CAc);
    enc_adaln_k<<<BSKn, 128, 0, stream>>>(f_x, q2k, f_cs, f_cb, f_kn);
    // q / k / v
    enc_gemm<0,1,4><<<dim3(BSQn/128, 2), 128, 0, stream>>>(f_qn, wq + b*WM,
        bq + b*CAc, nullptr, f_q, BSQn, CAc, CAc);
    enc_gemm<0,0,4><<<dim3(BSKn/128, 2), 128, 0, stream>>>(f_kn, wk + b*WM,
        nullptr, nullptr, f_k, BSKn, CAc, CAc);
    enc_gemm<0,0,4><<<dim3(BSKn/128, 2), 128, 0, stream>>>(f_kn, wv + b*WM,
        nullptr, nullptr, f_v, BSKn, CAc, CAc);
    // attention core
    enc_attn<<<dim3(SS, NHc, BBc), 128, 0, stream>>>(f_q, f_k, f_v, f_pl, o_kmask,
        f_o, b);
    // g = sigmoid(qn@wg) (reuse f_q), o *= g, o2 = o@wo
    enc_gemm<0,2,4><<<dim3(BSQn/128, 2), 128, 0, stream>>>(f_qn, wg + b*WM,
        nullptr, nullptr, f_q, BSQn, CAc, CAc);
    enc_mul_inplace<<<g1(BSQn*CAc, TH), TH, 0, stream>>>(f_o, f_q, BSQn*CAc);
    enc_gemm<0,0,4><<<dim3(BSQn/128, 2), 128, 0, stream>>>(f_o, wo + b*WM,
        nullptr, nullptr, f_o2, BSQn, CAc, CAc);
    // z-gate and residual
    enc_gemm<2,2,4><<<dim3(SQn/128, 2), 128, 0, stream>>>(o_qsc, wz + b*WM,
        bz + b*CAc, zln + b*CAc, f_gate, SQn, CAc, CAc);
    enc_addgate<<<g1(BSQn*CAc, TH), TH, 0, stream>>>(f_x, f_o2, f_gate,
        BSQn*CAc, SQn*CAc);
    // transition: tn = adaLN(x)
    enc_gemm<2,2,4><<<dim3(SQn/128, 2), 128, 0, stream>>>(o_qsc, t_ws + b*WM,
        t_bs + b*CAc, tln + b*CAc, f_cs, SQn, CAc, CAc);
    enc_gemm<2,0,4><<<dim3(SQn/128, 2), 128, 0, stream>>>(o_qsc, t_wb + b*WM,
        nullptr, tln + b*CAc, f_cb, SQn, CAc, CAc);
    enc_adaln_q<<<BSQn, 128, 0, stream>>>(f_x, f_cs, f_cb, f_qn);
    // a = tn @ w1 (128->512) in f_k; h = swish(a1)*a2 in f_v; h2 = h@w2 in f_o
    enc_gemm<0,0,4><<<dim3(BSQn/128, 8), 128, 0, stream>>>(f_qn, w1 + (size_t)b*CAc*512,
        nullptr, nullptr, f_k, BSQn, CAc, 512);
    enc_swiglu<<<g1(BSQn*256, TH), TH, 0, stream>>>(f_k, f_v, BSQn*256);
    enc_gemm<0,0,4><<<dim3(BSQn/128, 2), 128, 0, stream>>>(f_v, w2 + (size_t)b*256*CAc,
        nullptr, nullptr, f_o, BSQn, 256, CAc);
    enc_gemm<2,2,4><<<dim3(SQn/128, 2), 128, 0, stream>>>(o_qsc, twz + b*WM,
        tbz + b*CAc, tzln + b*CAc, f_gate, SQn, CAc, CAc);
    enc_addgate<<<g1(BSQn*CAc, TH), TH, 0, stream>>>(f_x, f_o, f_gate,
        BSQn*CAc, SQn*CAc);
  }

  // 10) final mask + skip output
  enc_mask_skip<<<g1(BSQn*CAc, TH), TH, 0, stream>>>(f_x, o_qmask, o_skip, BSQn*CAc);

  // 11) feat = x @ w_aggr (128->768) into f_k (+f_v) region
  enc_gemm<0,0,4><<<dim3(BSQn/128, CTc/64), 128, 0, stream>>>(f_x, w_aggr,
      nullptr, nullptr, f_k, BSQn, CAc, CTc);

  // 12) token aggregation
  enc_token_aggr<<<dim3(BBc*TT), 256, 0, stream>>>(f_k, atom_mask, o_tok);
}
